// STformer_82394652606787
// MI455X (gfx1250) — matrix operand order repaired and checked by reading, not compiled, not run on hardware
//
#include <hip/hip_runtime.h>
#include <cstddef>

// ---------------- types ----------------
typedef _Float16 f16x8  __attribute__((ext_vector_type(8)));
typedef _Float16 f16x16 __attribute__((ext_vector_type(16)));
typedef float    f32x8  __attribute__((ext_vector_type(8)));

#define EPS 1e-5f
#define TPOOL 82
#define NB 32
#define MROWS (NB * TPOOL)   // 2624

// region (spatial conv) constants, 0-based electrodes
__constant__ int c_nr[5]  = {4, 7, 9, 7, 7};
__constant__ int c_off[5] = {0, 4, 11, 20, 27};
__constant__ int c_el[34] = {0,2,3,4,
                             1,2,6,7,8,13,14,
                             2,3,4,8,9,10,14,15,16,
                             4,5,10,11,12,16,17,
                             14,15,16,18,19,20,21};

// ---------------- helpers ----------------
__device__ __forceinline__ float eluf(float x) { return x > 0.f ? x : (expf(x) - 1.f); }

__device__ __forceinline__ float wred(float v) {
#pragma unroll
  for (int o = 16; o > 0; o >>= 1) v += __shfl_xor(v, o, 32);
  return v;
}

__device__ __forceinline__ f32x8 zero8() {
  f32x8 z;
#pragma unroll
  for (int i = 0; i < 8; i++) z[i] = 0.f;
  return z;
}

// A-matrix 16x32 f16 fragment: lane<16 -> K base+0..7 and base+16..23 (base=kt),
// lane>=16 -> base=kt+8. Caller passes p = &A[row][kbase].
__device__ __forceinline__ f16x16 ld_frag_a(const _Float16* p) {
  f16x8 x0 = *(const f16x8*)p;
  f16x8 x1 = *(const f16x8*)(p + 16);
  f16x16 a;
#pragma unroll
  for (int i = 0; i < 8; i++) { a[i] = x0[i]; a[i + 8] = x1[i]; }
  return a;
}

// B-matrix 32x16 f16 fragment: lane<16 holds K kt+0..7 and kt+16..23 of column n, lane>=16 holds kt+8..15 and kt+24..31.
// Caller passes p = &W[n][kbase] (kbase=kt or kt+8; halves p[0..7] and p[16..23]).
__device__ __forceinline__ f16x16 ld_frag_b(const _Float16* p) {
  f16x8 x0 = *(const f16x8*)p;
  f16x8 x1 = *(const f16x8*)(p + 16);
  f16x16 b;
#pragma unroll
  for (int i = 0; i < 8; i++) { b[i] = x0[i]; b[i + 8] = x1[i]; }
  return b;
}

// ---------------- weight prep: f32 [N][K] -> f16 [Npad][Kp] (+ padded bias) ----------------
__global__ void prep_weight(const float* __restrict__ src, const float* __restrict__ bsrc,
                            _Float16* __restrict__ dst, float* __restrict__ bdst,
                            int N, int K, int Kp, int Npad) {
  int idx = blockIdx.x * blockDim.x + threadIdx.x;
  int total = Npad * Kp;
  if (idx < total) {
    int r = idx / Kp, c = idx % Kp;
    float v = (r < N && c < K) ? src[r * K + c] : 0.f;
    dst[idx] = (_Float16)v;
  }
  if (bdst && idx < Npad) bdst[idx] = (idx < N) ? bsrc[idx] : 0.f;
}

// ---------------- fused front end ----------------
struct FrontParams {
  const float* x;
  const float* sw[5]; const float* sb[5];
  const float* b1g; const float* b1b; const float* b1m; const float* b1v;
  const float* dww; const float* dwb;
  const _Float16* Wpw;       // f16 [128][608]
  const float* pwb;
  const float* b2g; const float* b2b; const float* b2m; const float* b2v;
  float* hout;               // f32 [32][120][82]
};

__global__ __launch_bounds__(256) void front_kernel(FrontParams P) {
  __shared__ _Float16 xs[22][72];
  __shared__ _Float16 sfh[12][72];
  __shared__ _Float16 Amat[48][616];   // M=48 (time), K padded 600->616 storage (608 used)
  const int tid = threadIdx.x;
  const int tp = blockIdx.x, b = blockIdx.y;
  const int t0 = tp * 48;

  for (int i = tid; i < 22 * 71; i += 256) {
    int c = i / 71, tl = i % 71;
    xs[c][tl] = (_Float16)P.x[(size_t)b * 88000 + c * 4000 + t0 + tl];
  }
  for (int i = tid; i < 48 * 16; i += 256) Amat[i / 16][600 + (i & 15)] = (_Float16)0.f;
  __syncthreads();

  // spatial conv + bn1 + depthwise(1,24), processed in k-groups of 12 rows
  for (int g = 0; g < 50; g++) {
    for (int i = tid; i < 12 * 71; i += 256) {
      int kk = i / 71, tl = i % 71;
      int k = g * 12 + kk;
      int f = k / 5, r = k % 5;
      int nr = c_nr[r], off = c_off[r];
      float s = 0.f;
      for (int e = 0; e < nr; e++) s += (float)xs[c_el[off + e]][tl] * P.sw[r][f * nr + e];
      float sc = P.b1g[f] * rsqrtf(P.b1v[f] + EPS);
      sfh[kk][tl] = (_Float16)((s + P.sb[r][f] - P.b1m[f]) * sc + P.b1b[f]);
    }
    __syncthreads();
    for (int i = tid; i < 12 * 48; i += 256) {
      int kk = i / 48, t = i % 48;
      int k = g * 12 + kk;
      int f = k / 5;
      float a = P.dwb[f];
#pragma unroll
      for (int j = 0; j < 24; j++) a += (float)sfh[kk][t + j] * P.dww[f * 24 + j];
      Amat[t][k] = (_Float16)a;
    }
    __syncthreads();
  }

  // pointwise GEMM: M=48 (3 tiles), N=128 (8 waves x 1 tile), K=608 (19 tiles)
  const int wave = tid >> 5, lane = tid & 31;
  const int n = wave * 16 + (lane & 15);
  f32x8 acc[3];
  for (int i = 0; i < 3; i++) acc[i] = zero8();
  for (int kt = 0; kt < 19; kt++) {
    f16x16 bf = ld_frag_b(P.Wpw + (size_t)n * 608 + kt * 32 + ((lane & 16) ? 8 : 0));
#pragma unroll
    for (int mi = 0; mi < 3; mi++) {
      f16x16 af = ld_frag_a(&Amat[mi * 16 + (lane & 15)][kt * 32 + ((lane & 16) ? 8 : 0)]);
      acc[mi] = __builtin_amdgcn_wmma_f32_16x16x32_f16(false, af, false, bf, (short)0,
                                                       acc[mi], false, false);
    }
  }

  // epilogue: +bias, bn2, elu, variance pool over the 48 time steps
  float bv = 0.f, sc2 = 0.f, sh2 = 0.f;
  if (n < 120) {
    bv  = P.pwb[n];
    sc2 = P.b2g[n] * rsqrtf(P.b2v[n] + EPS);
    sh2 = P.b2b[n] - P.b2m[n] * sc2;
  }
  float s1 = 0.f, s2 = 0.f;
#pragma unroll
  for (int mi = 0; mi < 3; mi++) {
#pragma unroll
    for (int i = 0; i < 8; i++) {
      float y = (acc[mi][i] + bv) * sc2 + sh2;
      float e = eluf(y);
      s1 += e; s2 += e * e;
    }
  }
  float s1p = __shfl_xor(s1, 16, 32);
  float s2p = __shfl_xor(s2, 16, 32);
  if (lane < 16 && n < 120) {
    float m1 = (s1 + s1p) / 48.f;
    float m2 = (s2 + s2p) / 48.f;
    P.hout[(size_t)b * 9840 + n * 82 + tp] = m2 - m1 * m1;
  }
}

// ---------------- generic WMMA GEMM: C[m][n] = sum_k A[m][k]*W[n][k] + bias[n] ----------------
// A f16 [M][Kp] (Kp multiple of 128), W f16 [Np][Kp], bias f32 [Np].
// grid = (ceil(M/64), Np/128); out f16 or f32 with row stride Np.
__global__ __launch_bounds__(256) void gemm_kernel(const _Float16* __restrict__ A,
                                                   const _Float16* __restrict__ W,
                                                   const float* __restrict__ bias,
                                                   void* __restrict__ Cout,
                                                   int M, int Kp, int Np, int outF16) {
  __shared__ _Float16 As[64][136];
  const int tid = threadIdx.x;
  const int wave = tid >> 5, lane = tid & 31;
  const int m0 = blockIdx.x * 64;
  const int n0 = blockIdx.y * 128 + wave * 16;
  f32x8 acc[4];
  for (int i = 0; i < 4; i++) acc[i] = zero8();

  for (int kc = 0; kc < Kp; kc += 128) {
    for (int i = tid; i < 64 * 16; i += 256) {
      int r = i >> 4, c8 = i & 15;
      int m = m0 + r;
      f16x8 v;
      if (m < M) v = *(const f16x8*)(A + (size_t)m * Kp + kc + c8 * 8);
      else { for (int q = 0; q < 8; q++) v[q] = (_Float16)0.f; }
      *(f16x8*)(&As[r][c8 * 8]) = v;
    }
    {   // prefetch next A chunk (global_prefetch_b8)
      int mpf = m0 + (tid & 63);
      if (kc + 128 < Kp && mpf < M)
        __builtin_prefetch((const void*)(A + (size_t)mpf * Kp + kc + 128), 0, 1);
    }
    __syncthreads();
#pragma unroll
    for (int kt = 0; kt < 128; kt += 32) {
      f16x16 bf = ld_frag_b(W + (size_t)(n0 + (lane & 15)) * Kp + kc + kt + ((lane & 16) ? 8 : 0));
#pragma unroll
      for (int mi = 0; mi < 4; mi++) {
        f16x16 af = ld_frag_a(&As[mi * 16 + (lane & 15)][kt + ((lane & 16) ? 8 : 0)]);
        acc[mi] = __builtin_amdgcn_wmma_f32_16x16x32_f16(false, af, false, bf, (short)0,
                                                         acc[mi], false, false);
      }
    }
    __syncthreads();
  }

  const int n = n0 + (lane & 15);
  const float bv = bias[n];
#pragma unroll
  for (int mi = 0; mi < 4; mi++) {
    int mb = m0 + mi * 16 + ((lane & 16) ? 8 : 0);
#pragma unroll
    for (int i = 0; i < 8; i++) {
      int m = mb + i;
      if (m < M) {
        float y = acc[mi][i] + bv;
        if (outF16) ((_Float16*)Cout)[(size_t)m * Np + n] = (_Float16)y;
        else        ((float*)Cout)[(size_t)m * Np + n] = y;
      }
    }
  }
}

// ---------------- layernorm over channels (transformer input) ----------------
__global__ __launch_bounds__(256) void ln0_kernel(const float* __restrict__ h,
                                                  const float* __restrict__ g,
                                                  const float* __restrict__ bb,
                                                  float* __restrict__ t_f32,
                                                  _Float16* __restrict__ tln) {
  const int wave = threadIdx.x >> 5, lane = threadIdx.x & 31;
  const int m = blockIdx.x * 8 + wave;
  const int b = m / 82, tp = m % 82;
  float v[4]; float s = 0.f;
#pragma unroll
  for (int c = 0; c < 4; c++) {
    int j = lane + c * 32;
    v[c] = (j < 120) ? h[(size_t)b * 9840 + j * 82 + tp] : 0.f;
    s += v[c];
  }
  s = wred(s);
  float mu = s / 120.f;
  float q = 0.f;
#pragma unroll
  for (int c = 0; c < 4; c++) { int j = lane + c * 32; float d = (j < 120) ? v[c] - mu : 0.f; q += d * d; }
  q = wred(q);
  float rs = rsqrtf(q / 120.f + EPS);
#pragma unroll
  for (int c = 0; c < 4; c++) {
    int j = lane + c * 32;
    if (j < 120) {
      float tv = (v[c] - mu) * rs * g[j] + bb[j];
      t_f32[(size_t)m * 120 + j] = tv;
      tln[(size_t)m * 128 + j] = (_Float16)tv;
    } else {
      tln[(size_t)m * 128 + j] = (_Float16)0.f;
    }
  }
}

// ---------------- attention (one block per (head,batch)) ----------------
__global__ __launch_bounds__(256) void attn_kernel(const _Float16* __restrict__ qkv,
                                                   _Float16* __restrict__ obuf) {
  const int h = blockIdx.x, b = blockIdx.y;
  __shared__ float Q[82][24], Kk[82][24], V[82][24];
  __shared__ float S[82][84];
  const int tid = threadIdx.x;
  const float scale = 0.20412414523193154f; // 1/sqrt(24)
  for (int i = tid; i < 82 * 24; i += 256) {
    int t = i / 24, d = i % 24;
    size_t row = (size_t)(b * 82 + t) * 384;
    Q[t][d]  = (float)qkv[row + h * 24 + d];
    Kk[t][d] = (float)qkv[row + 128 + h * 24 + d];
    V[t][d]  = (float)qkv[row + 256 + h * 24 + d];
  }
  __syncthreads();
  for (int i = tid; i < 82 * 82; i += 256) {
    int r = i / 82, c = i % 82;
    float s = 0.f;
#pragma unroll
    for (int d = 0; d < 24; d++) s += Q[r][d] * Kk[c][d];
    S[r][c] = s * scale;
  }
  __syncthreads();
  if (tid < 82) {
    float mx = -1e30f;
    for (int c = 0; c < 82; c++) mx = fmaxf(mx, S[tid][c]);
    float sum = 0.f;
    for (int c = 0; c < 82; c++) { float e = expf(S[tid][c] - mx); S[tid][c] = e; sum += e; }
    float inv = 1.f / sum;
    for (int c = 0; c < 82; c++) S[tid][c] *= inv;
  }
  __syncthreads();
  for (int i = tid; i < 82 * 24; i += 256) {
    int t = i / 24, d = i % 24;
    float o = 0.f;
    for (int j = 0; j < 82; j++) o += S[t][j] * V[j][d];
    obuf[(size_t)(b * 82 + t) * 128 + h * 24 + d] = (_Float16)o;
  }
  if (h == 0)
    for (int i = tid; i < 82 * 8; i += 256)
      obuf[(size_t)(b * 82 + i / 8) * 128 + 120 + (i & 7)] = (_Float16)0.f;
}

// ---------------- residual+LN1, ff(d_ff=1), residual+LN2, transpose-out ----------------
__global__ __launch_bounds__(256) void post_attn(const float* __restrict__ t_f32,
                                                 const float* __restrict__ g,
                                                 const float* ln1g, const float* ln1b,
                                                 const float* ff1w, const float* ff1b,
                                                 const float* ff2w, const float* ff2b,
                                                 const float* ln2g, const float* ln2b,
                                                 float* __restrict__ gbuf) {
  const int wave = threadIdx.x >> 5, lane = threadIdx.x & 31;
  const int m = blockIdx.x * 8 + wave;
  const int b = m / 82, tp = m % 82;
  float u[4], t1v[4], v2[4];
  float s = 0.f;
#pragma unroll
  for (int c = 0; c < 4; c++) {
    int j = lane + c * 32;
    u[c] = (j < 120) ? (t_f32[(size_t)m * 120 + j] + g[(size_t)m * 128 + j]) : 0.f;
    s += u[c];
  }
  s = wred(s); float mu = s / 120.f;
  float q = 0.f;
#pragma unroll
  for (int c = 0; c < 4; c++) { int j = lane + c * 32; float d = (j < 120) ? u[c] - mu : 0.f; q += d * d; }
  q = wred(q); float rs = rsqrtf(q / 120.f + EPS);
  float ffs = 0.f;
#pragma unroll
  for (int c = 0; c < 4; c++) {
    int j = lane + c * 32;
    if (j < 120) { t1v[c] = (u[c] - mu) * rs * ln1g[j] + ln1b[j]; ffs += t1v[c] * ff1w[j]; }
    else t1v[c] = 0.f;
  }
  ffs = wred(ffs) + ff1b[0];
  float e = eluf(ffs);
  float s2 = 0.f;
#pragma unroll
  for (int c = 0; c < 4; c++) {
    int j = lane + c * 32;
    v2[c] = (j < 120) ? (t1v[c] + e * ff2w[j] + ff2b[j]) : 0.f;
    s2 += v2[c];
  }
  s2 = wred(s2); float mu2 = s2 / 120.f;
  float q2 = 0.f;
#pragma unroll
  for (int c = 0; c < 4; c++) { int j = lane + c * 32; float d = (j < 120) ? v2[c] - mu2 : 0.f; q2 += d * d; }
  q2 = wred(q2); float rs2 = rsqrtf(q2 / 120.f + EPS);
#pragma unroll
  for (int c = 0; c < 4; c++) {
    int j = lane + c * 32;
    if (j < 120) gbuf[(size_t)b * 9840 + j * 82 + tp] = (v2[c] - mu2) * rs2 * ln2g[j] + ln2b[j];
  }
}

// ---------------- local branch depthwise(1,4) pad(1,2) ----------------
__global__ void local_dw(const float* __restrict__ h, const float* __restrict__ w,
                         const float* __restrict__ bb, _Float16* __restrict__ Aout) {
  int idx = blockIdx.x * 256 + threadIdx.x;
  if (idx >= MROWS * 128) return;
  int m = idx >> 7, f = idx & 127;
  if (f >= 120) { Aout[idx] = (_Float16)0.f; return; }
  int b = m / 82, tp = m % 82;
  const float* hp = h + (size_t)b * 9840 + f * 82;
  float acc = bb[f];
#pragma unroll
  for (int j = 0; j < 4; j++) {
    int tt = tp - 1 + j;
    if (tt >= 0 && tt < 82) acc += hp[tt] * w[f * 4 + j];
  }
  Aout[idx] = (_Float16)acc;
}

// ---------------- combine: h_next = global + elu(bn(local_pw)) ----------------
__global__ void combine_kernel(const float* __restrict__ gbuf, const float* __restrict__ g2,
                               const float* gg, const float* bb, const float* mm, const float* vv,
                               float* __restrict__ hout) {
  int idx = blockIdx.x * 256 + threadIdx.x;
  if (idx >= 32 * 9840) return;
  int b = idx / 9840, r = idx % 9840;
  int f = r / 82, tp = r % 82;
  int m = b * 82 + tp;
  float sc = gg[f] * rsqrtf(vv[f] + EPS);
  float y = (g2[(size_t)m * 128 + f] - mm[f]) * sc + bb[f];
  hout[idx] = gbuf[idx] + eluf(y);
}

// ---------------- flatten + cast features for classifier l1 ----------------
__global__ void cast_pad(const float* __restrict__ h, _Float16* __restrict__ out) {
  int idx = blockIdx.x * 256 + threadIdx.x;
  if (idx >= 32 * 9856) return;
  int b = idx / 9856, c = idx % 9856;
  out[idx] = (_Float16)((c < 9840) ? h[(size_t)b * 9840 + c] : 0.f);
}

// ---------------- classifier tail: bn+elu, l2, bn+elu, l3, log_softmax ----------------
__global__ __launch_bounds__(256) void cls_tail(const float* __restrict__ l1out,
    const float* bg1, const float* bb1, const float* bm1, const float* bv1,
    const float* w2, const float* b2,
    const float* bg2, const float* bb2, const float* bm2, const float* bv2,
    const float* w3, const float* b3, float* __restrict__ out) {
  const int b = blockIdx.x, tid = threadIdx.x;
  __shared__ float f1[256];
  __shared__ float f2[64];
  float v = l1out[b * 256 + tid];
  v = (v - bm1[tid]) * (bg1[tid] * rsqrtf(bv1[tid] + EPS)) + bb1[tid];
  f1[tid] = eluf(v);
  __syncthreads();
  if (tid < 64) {
    float a = b2[tid];
    for (int i = 0; i < 256; i++) a += f1[i] * w2[tid * 256 + i];
    a = (a - bm2[tid]) * (bg2[tid] * rsqrtf(bv2[tid] + EPS)) + bb2[tid];
    f2[tid] = eluf(a);
  }
  __syncthreads();
  if (tid == 0) {
    float lg[4]; float mx = -1e30f;
    for (int o = 0; o < 4; o++) {
      float a = b3[o];
      for (int i = 0; i < 64; i++) a += f2[i] * w3[o * 64 + i];
      lg[o] = a; mx = fmaxf(mx, a);
    }
    float sum = 0.f;
    for (int o = 0; o < 4; o++) sum += expf(lg[o] - mx);
    float lse = logf(sum) + mx;
    for (int o = 0; o < 4; o++) out[b * 4 + o] = lg[o] - lse;
  }
}

// ---------------- host ----------------
extern "C" void kernel_launch(void* const* d_in, const int* in_sizes, int n_in,
                              void* d_out, int out_size, void* d_ws, size_t ws_size,
                              hipStream_t stream) {
  (void)in_sizes; (void)n_in; (void)out_size; (void)ws_size;
  auto F = [&](int i) { return (const float*)d_in[i]; };
  char* ws = (char*)d_ws;
  size_t off = 0;
  auto alloc = [&](size_t sz) { off = (off + 255) & ~(size_t)255; size_t o = off; off += sz; return o; };

  size_t oWpw = alloc(128 * 608 * 2);
  size_t oWqkv[3], oBqkv[3], oWo[3], oBo[3], oWlpw[3], oBlpw[3];
  for (int l = 0; l < 3; l++) {
    oWqkv[l] = alloc(384 * 128 * 2); oBqkv[l] = alloc(384 * 4);
    oWo[l]   = alloc(128 * 128 * 2); oBo[l]   = alloc(128 * 4);
    oWlpw[l] = alloc(128 * 128 * 2); oBlpw[l] = alloc(128 * 4);
  }
  size_t oWl1 = alloc((size_t)256 * 9856 * 2), oBl1 = alloc(256 * 4);
  size_t oHA = alloc((size_t)32 * 9840 * 4), oHB = alloc((size_t)32 * 9840 * 4);
  size_t oT   = alloc((size_t)MROWS * 120 * 4);
  size_t oTln = alloc((size_t)MROWS * 128 * 2);
  size_t oQKV = alloc((size_t)MROWS * 384 * 2);
  size_t oO   = alloc((size_t)MROWS * 128 * 2);
  size_t oG   = alloc((size_t)MROWS * 128 * 4);
  size_t oGB  = alloc((size_t)32 * 9840 * 4);
  size_t oAl  = alloc((size_t)MROWS * 128 * 2);
  size_t oG2  = alloc((size_t)MROWS * 128 * 4);
  size_t oFe  = alloc((size_t)32 * 9856 * 2);
  size_t oL1o = alloc((size_t)32 * 256 * 4);

  auto H16 = [&](size_t o) { return (_Float16*)(ws + o); };
  auto F32 = [&](size_t o) { return (float*)(ws + o); };

  auto prep = [&](const float* w, const float* bsrc, _Float16* wdst, float* bdst,
                  int N, int K, int Kp, int Npad) {
    int total = Npad * Kp;
    prep_weight<<<(total + 255) / 256, 256, 0, stream>>>(w, bsrc, wdst, bdst, N, K, Kp, Npad);
  };

  // weight prep (f16, transposed-for-B layout = natural [out][in], zero padded)
  prep(F(17), nullptr, H16(oWpw), nullptr, 120, 600, 608, 128);
  for (int l = 0; l < 3; l++) {
    int lb = 23 + l * 26;
    prep(F(lb + 2), F(lb + 3), H16(oWqkv[l]),                 F32(oBqkv[l]),       120, 120, 128, 128);
    prep(F(lb + 4), F(lb + 5), H16(oWqkv[l]) + 128 * 128,     F32(oBqkv[l]) + 128, 120, 120, 128, 128);
    prep(F(lb + 6), F(lb + 7), H16(oWqkv[l]) + 256 * 128,     F32(oBqkv[l]) + 256, 120, 120, 128, 128);
    prep(F(lb + 8), F(lb + 9), H16(oWo[l]),   F32(oBo[l]),   120, 120, 128, 128);
    prep(F(lb + 20), F(lb + 21), H16(oWlpw[l]), F32(oBlpw[l]), 120, 120, 128, 128);
  }
  prep(F(101), F(102), H16(oWl1), F32(oBl1), 256, 9840, 9856, 256);

  // fused front end
  FrontParams FP;
  FP.x = F(0);
  for (int r = 0; r < 5; r++) { FP.sw[r] = F(1 + 2 * r); FP.sb[r] = F(2 + 2 * r); }
  FP.b1g = F(11); FP.b1b = F(12); FP.b1m = F(13); FP.b1v = F(14);
  FP.dww = F(15); FP.dwb = F(16);
  FP.Wpw = H16(oWpw); FP.pwb = F(18);
  FP.b2g = F(19); FP.b2b = F(20); FP.b2m = F(21); FP.b2v = F(22);
  FP.hout = F32(oHA);
  front_kernel<<<dim3(TPOOL, NB), 256, 0, stream>>>(FP);

  float* hcur = F32(oHA);
  float* hnext = F32(oHB);
  for (int l = 0; l < 3; l++) {
    int lb = 23 + l * 26;
    ln0_kernel<<<MROWS / 8, 256, 0, stream>>>(hcur, F(lb + 0), F(lb + 1), F32(oT), H16(oTln));
    gemm_kernel<<<dim3(41, 3), 256, 0, stream>>>(H16(oTln), H16(oWqkv[l]), F32(oBqkv[l]),
                                                 (void*)H16(oQKV), MROWS, 128, 384, 1);
    attn_kernel<<<dim3(5, NB), 256, 0, stream>>>(H16(oQKV), H16(oO));
    gemm_kernel<<<dim3(41, 1), 256, 0, stream>>>(H16(oO), H16(oWo[l]), F32(oBo[l]),
                                                 (void*)F32(oG), MROWS, 128, 128, 0);
    post_attn<<<MROWS / 8, 256, 0, stream>>>(F32(oT), F32(oG),
        F(lb + 10), F(lb + 11), F(lb + 14), F(lb + 15), F(lb + 16), F(lb + 17),
        F(lb + 12), F(lb + 13), F32(oGB));
    local_dw<<<(MROWS * 128) / 256, 256, 0, stream>>>(hcur, F(lb + 18), F(lb + 19), H16(oAl));
    gemm_kernel<<<dim3(41, 1), 256, 0, stream>>>(H16(oAl), H16(oWlpw[l]), F32(oBlpw[l]),
                                                 (void*)F32(oG2), MROWS, 128, 128, 0);
    combine_kernel<<<(32 * 9840 + 255) / 256, 256, 0, stream>>>(F32(oGB), F32(oG2),
        F(lb + 22), F(lb + 23), F(lb + 24), F(lb + 25), hnext);
    float* tmp = hcur; hcur = hnext; hnext = tmp;
  }

  cast_pad<<<(32 * 9856) / 256, 256, 0, stream>>>(hcur, H16(oFe));
  gemm_kernel<<<dim3(1, 2), 256, 0, stream>>>(H16(oFe), H16(oWl1), F32(oBl1),
                                              (void*)F32(oL1o), 32, 9856, 256, 0);
  cls_tail<<<NB, 256, 0, stream>>>(F32(oL1o),
      F(103), F(104), F(105), F(106), F(107), F(108),
      F(109), F(110), F(111), F(112), F(113), F(114), (float*)d_out);
}
